// MultiHeadAttention_40862318854775
// MI455X (gfx1250) — matrix operand order repaired and checked by reading, not compiled, not run on hardware
//
#include <hip/hip_runtime.h>
#include <stdint.h>

#define B_  2
#define S_  2048
#define D_  2048
#define H_  16
#define HD_ 128
#define N_  (B_*S_)   // 4096 rows

typedef __attribute__((ext_vector_type(16))) __bf16        v16bf;
typedef __attribute__((ext_vector_type(8)))  float         v8f;
typedef __attribute__((ext_vector_type(4)))  float         v4f;
typedef __attribute__((ext_vector_type(4)))  unsigned int  v4u;
typedef __attribute__((ext_vector_type(2)))  unsigned int  v2u;

union Frag16 { v16bf v; v4u q[2]; };
union Acc8   { v8f v; float f[8]; };

static __device__ __forceinline__ unsigned short f2bf(float x) {
  unsigned int u = __float_as_uint(x);
  u += 0x7fffu + ((u >> 16) & 1u);          // round-to-nearest-even
  return (unsigned short)(u >> 16);
}
static __device__ __forceinline__ unsigned int pk2(float a, float b) {
  return (unsigned int)f2bf(a) | ((unsigned int)f2bf(b) << 16);
}

// ---- CDNA5 async global->LDS DMA (ASYNCcnt-tracked), per ISA ch.10 / 15.18.3
static __device__ __forceinline__ void async_ldg_b128(unsigned int lds_off,
                                                      const void* g) {
  asm volatile("global_load_async_to_lds_b128 %0, %1, off"
               :: "v"(lds_off), "v"((unsigned long long)g)
               : "memory");
}
template <int N>
static __device__ __forceinline__ void wait_async() {
  asm volatile("s_wait_asynccnt %0" :: "i"(N) : "memory");
}

// ---------------------------------------------------------------- fp32 -> bf16
__global__ void cvt_f32_bf16(const float* __restrict__ src,
                             unsigned short* __restrict__ dst, int n4) {
  int i = blockIdx.x * blockDim.x + threadIdx.x;
  if (i >= n4) return;
  const float* s = src + (size_t)i * 4;
  v2u p;
  p.x = pk2(s[0], s[1]);
  p.y = pk2(s[2], s[3]);
  *(v2u*)&dst[(size_t)i * 4] = p;
}

// ---------------------------------------------------------------- WMMA GEMM
// C(row,col) = sum_k A(row,k) * W(col,k) + bias(col)   (i.e. X @ W^T + b)
// MODE 0: store bf16 head-split (B,H,S,HD)
// MODE 1: store bf16 transposed head-split (B,H,HD,S)   (for V^T)
// MODE 2: store fp32 row-major (final output)
// ABF16: A operand already bf16; else raw fp32 staged and converted per-frag
template <int MODE, bool ABF16>
__global__ __launch_bounds__(256) void gemm_wmma(
    const float* __restrict__ Xf, const unsigned short* __restrict__ Xb,
    const unsigned short* __restrict__ Wb, const float* __restrict__ bias,
    unsigned short* __restrict__ outb, float* __restrict__ outf) {
  // double-buffered tiles; A kept in its source precision
  __shared__ unsigned char   AsRaw[2][ABF16 ? 128 * 32 * 2 : 128 * 32 * 4];
  __shared__ unsigned short  Bs[2][128 * 32];   // [col][k] bf16
  const int tid  = threadIdx.x;
  const int lane = tid & 31;
  const int wave = tid >> 5;
  const int waveM = wave >> 2;              // 0..1  -> 64 rows each
  const int waveN = wave & 3;               // 0..3  -> 32 cols each
  const int bM = blockIdx.x * 128;
  const int bN = blockIdx.y * 128;
  const int m  = lane & 15;
  const int hf = lane >> 4;
  constexpr int OPS = ABF16 ? 4 : 6;        // async b128 ops per thread per tile

  Acc8 acc[4][2];
#pragma unroll
  for (int i = 0; i < 4; ++i)
#pragma unroll
    for (int j = 0; j < 2; ++j)
#pragma unroll
      for (int r = 0; r < 8; ++r) acc[i][j].f[r] = 0.f;

  auto stage = [&](int k0, int buf) {
#pragma unroll
    for (int i = 0; i < 2; ++i) {
      int id  = tid * 2 + i;                // 512 runs of 8 elements
      int row = id >> 2;
      int kr  = (id & 3) * 8;
      if (ABF16) {
        async_ldg_b128((unsigned int)(size_t)&AsRaw[buf][(row * 32 + kr) * 2],
                       &Xb[(size_t)(bM + row) * D_ + k0 + kr]);
      } else {
        const float* g = &Xf[(size_t)(bM + row) * D_ + k0 + kr];
        unsigned int lo = (unsigned int)(size_t)&AsRaw[buf][(row * 32 + kr) * 4];
        async_ldg_b128(lo, g);
        async_ldg_b128(lo + 16, g + 4);
      }
      async_ldg_b128((unsigned int)(size_t)&Bs[buf][row * 32 + kr],
                     &Wb[(size_t)(bN + row) * D_ + k0 + kr]);
    }
  };

  stage(0, 0);
  const int T = D_ / 32;
  for (int kt = 0; kt < T; ++kt) {
    const int cur = kt & 1;
    if (kt + 1 < T) {
      stage((kt + 1) * 32, cur ^ 1);
      wait_async<OPS>();                    // tile kt landed; kt+1 in flight
    } else {
      wait_async<0>();
    }
    __syncthreads();

    // A fragment (16x32 bf16): lane m holds K runs [hf*8,+8) and [16+hf*8,+8)
    Frag16 af[4], bfr[2];
    if (ABF16) {
      const unsigned short* A = (const unsigned short*)AsRaw[cur];
#pragma unroll
      for (int fm = 0; fm < 4; ++fm) {
        int base = (waveM * 64 + fm * 16 + m) * 32;
        af[fm].q[0] = *(const v4u*)&A[base + hf * 8];
        af[fm].q[1] = *(const v4u*)&A[base + 16 + hf * 8];
      }
    } else {
      const float* A = (const float*)AsRaw[cur];
#pragma unroll
      for (int fm = 0; fm < 4; ++fm) {
        int base = (waveM * 64 + fm * 16 + m) * 32;
        v4f a0 = *(const v4f*)&A[base + hf * 8];
        v4f a1 = *(const v4f*)&A[base + hf * 8 + 4];
        v4f a2 = *(const v4f*)&A[base + 16 + hf * 8];
        v4f a3 = *(const v4f*)&A[base + 16 + hf * 8 + 4];
        v4u p0 = {pk2(a0.x, a0.y), pk2(a0.z, a0.w), pk2(a1.x, a1.y), pk2(a1.z, a1.w)};
        v4u p1 = {pk2(a2.x, a2.y), pk2(a2.z, a2.w), pk2(a3.x, a3.y), pk2(a3.z, a3.w)};
        af[fm].q[0] = p0;
        af[fm].q[1] = p1;
      }
    }
    // B fragment (32x16 bf16): lane m = column, K runs [hf*8,+8) and [16+hf*8,+8)
#pragma unroll
    for (int fn = 0; fn < 2; ++fn) {
      int base = (waveN * 32 + fn * 16 + m) * 32 + hf * 8;
      bfr[fn].q[0] = *(const v4u*)&Bs[cur][base];
      bfr[fn].q[1] = *(const v4u*)&Bs[cur][base + 16];
    }
#pragma unroll
    for (int fm = 0; fm < 4; ++fm)
#pragma unroll
      for (int fn = 0; fn < 2; ++fn)
        acc[fm][fn].v = __builtin_amdgcn_wmma_f32_16x16x32_bf16(
            false, af[fm].v, false, bfr[fn].v, (short)0, acc[fm][fn].v,
            false, false);
    __syncthreads();                        // reads done before buffer reuse
  }

  // C layout: VGPR r -> row (r + 8*hf), lane m -> col
#pragma unroll
  for (int fm = 0; fm < 4; ++fm) {
#pragma unroll
    for (int fn = 0; fn < 2; ++fn) {
      int col = bN + waveN * 32 + fn * 16 + m;
      float bi = bias[col];
#pragma unroll
      for (int r = 0; r < 8; ++r) {
        int row = bM + waveM * 64 + fm * 16 + hf * 8 + r;
        float val = acc[fm][fn].f[r] + bi;
        if (MODE == 2) {
          outf[(size_t)row * D_ + col] = val;
        } else {
          int b = row >> 11, s = row & (S_ - 1);
          int h = col >> 7, hd = col & (HD_ - 1);
          size_t idx;
          if (MODE == 0)
            idx = (((size_t)(b * H_ + h)) * S_ + s) * HD_ + hd;
          else
            idx = (((size_t)(b * H_ + h)) * HD_ + hd) * S_ + s;
          outb[idx] = f2bf(val);
        }
      }
    }
  }
}

// ---------------------------------------------------------------- flash attn
// Q,K: (B,H,S,HD) bf16; Vt: (B,H,HD,S) bf16; O: (B,S,D) bf16
__global__ __launch_bounds__(256) void flash_wmma(
    const unsigned short* __restrict__ Q, const unsigned short* __restrict__ K,
    const unsigned short* __restrict__ Vt, unsigned short* __restrict__ O) {
  __shared__ unsigned short Ks[64 * 128];     // [key][hd]   16KB
  __shared__ unsigned short Vs[128 * 64];     // [hd][key]   16KB
  __shared__ unsigned short Ps[8][16 * 64];   // per-wave P  16KB
  const int tid = threadIdx.x, lane = tid & 31, wave = tid >> 5;
  const int m = lane & 15, hf = lane >> 4;
  const int bh = blockIdx.y;                  // b*H + h
  const int q0 = blockIdx.x * 128;
  const int qrow0 = q0 + wave * 16;           // this wave's 16 query rows
  const size_t baseQK = (size_t)bh * S_ * HD_;
  const size_t baseV  = (size_t)bh * HD_ * S_;

  // Q fragments: 16 rows x 128 hd = 4 A-frags (resident for whole kernel)
  Frag16 qf[4];
  const size_t qb = baseQK + (size_t)(qrow0 + m) * HD_;
#pragma unroll
  for (int kk = 0; kk < 4; ++kk) {
    qf[kk].q[0] = *(const v4u*)&Q[qb + kk * 32 + hf * 8];
    qf[kk].q[1] = *(const v4u*)&Q[qb + kk * 32 + 16 + hf * 8];
  }

  Acc8 o[8];
  float mr[8], lr[8];
#pragma unroll
  for (int d = 0; d < 8; ++d)
#pragma unroll
    for (int r = 0; r < 8; ++r) o[d].f[r] = 0.f;
#pragma unroll
  for (int r = 0; r < 8; ++r) { mr[r] = -1e30f; lr[r] = 0.f; }

  const float scl = 0.08838834764831845f;     // 1/sqrt(128)
  const int nkt = (q0 + 128) >> 6;            // 64-key tiles up to the diagonal

  for (int kt = 0; kt < nkt; ++kt) {
    const int kb = kt * 64;
    __syncthreads();                          // all waves done reading tiles
    // async-DMA K tile (64 keys x 128 hd) and V^T tile (128 hd x 64 keys)
#pragma unroll
    for (int i = 0; i < 4; ++i) {
      int id = tid * 4 + i;
      int key = id >> 4, hr = (id & 15) * 8;
      async_ldg_b128((unsigned int)(size_t)&Ks[key * 128 + hr],
                     &K[baseQK + (size_t)(kb + key) * HD_ + hr]);
    }
#pragma unroll
    for (int i = 0; i < 4; ++i) {
      int id = tid * 4 + i;
      int hd = id >> 3, kr = (id & 7) * 8;
      async_ldg_b128((unsigned int)(size_t)&Vs[hd * 64 + kr],
                     &Vt[baseV + (size_t)hd * S_ + kb + kr]);
    }
    wait_async<0>();
    __syncthreads();

    // S = Q K^T : hoist 4 K-fragments per K-step, 4 independent WMMA chains
    Acc8 sc[4];
#pragma unroll
    for (int n = 0; n < 4; ++n)
#pragma unroll
      for (int r = 0; r < 8; ++r) sc[n].f[r] = 0.f;
#pragma unroll
    for (int kk = 0; kk < 4; ++kk) {
      Frag16 kf[4];
#pragma unroll
      for (int n = 0; n < 4; ++n) {
        int base = (n * 16 + m) * 128 + kk * 32 + hf * 8;
        kf[n].q[0] = *(const v4u*)&Ks[base];
        kf[n].q[1] = *(const v4u*)&Ks[base + 16];
      }
#pragma unroll
      for (int n = 0; n < 4; ++n)
        sc[n].v = __builtin_amdgcn_wmma_f32_16x16x32_bf16(
            false, qf[kk].v, false, kf[n].v, (short)0, sc[n].v, false, false);
    }
    // scale + causal mask
#pragma unroll
    for (int n = 0; n < 4; ++n)
#pragma unroll
      for (int r = 0; r < 8; ++r) {
        int qg = qrow0 + hf * 8 + r;
        int kg = kb + n * 16 + m;
        float x = sc[n].f[r] * scl;
        sc[n].f[r] = (kg <= qg) ? x : -1e30f;
      }

    // online softmax: row max across 4 subtiles and the 16 lanes of the row
    float mn[8], fs[8];
#pragma unroll
    for (int r = 0; r < 8; ++r) {
      float mx = fmaxf(fmaxf(sc[0].f[r], sc[1].f[r]),
                       fmaxf(sc[2].f[r], sc[3].f[r]));
#pragma unroll
      for (int off = 8; off >= 1; off >>= 1)
        mx = fmaxf(mx, __shfl_xor(mx, off, 16));
      mn[r] = fmaxf(mr[r], mx);
      fs[r] = __expf(mr[r] - mn[r]);
      mr[r] = mn[r];
    }
    // P = exp(S - m_new); bf16 to per-wave LDS scratch (C->A relayout)
#pragma unroll
    for (int n = 0; n < 4; ++n)
#pragma unroll
      for (int r = 0; r < 8; ++r) {
        float p = __expf(sc[n].f[r] - mn[r]);
        sc[n].f[r] = p;
        Ps[wave][(hf * 8 + r) * 64 + n * 16 + m] = f2bf(p);
      }
    // l update
#pragma unroll
    for (int r = 0; r < 8; ++r) {
      float sm = sc[0].f[r] + sc[1].f[r] + sc[2].f[r] + sc[3].f[r];
#pragma unroll
      for (int off = 8; off >= 1; off >>= 1) sm += __shfl_xor(sm, off, 16);
      lr[r] = lr[r] * fs[r] + sm;
    }
    // rescale O accumulators
#pragma unroll
    for (int d = 0; d < 8; ++d)
#pragma unroll
      for (int r = 0; r < 8; ++r) o[d].f[r] *= fs[r];

    // O += P @ V : hoist 4 V-fragments per group, independent WMMA chains
#pragma unroll
    for (int kk = 0; kk < 2; ++kk) {
      Frag16 pa;
      int pbp = m * 64 + kk * 32;
      pa.q[0] = *(const v4u*)&Ps[wave][pbp + hf * 8];
      pa.q[1] = *(const v4u*)&Ps[wave][pbp + 16 + hf * 8];
#pragma unroll
      for (int dg = 0; dg < 2; ++dg) {
        Frag16 vf[4];
#pragma unroll
        for (int d4 = 0; d4 < 4; ++d4) {
          int d = dg * 4 + d4;
          int vb = (d * 16 + m) * 64 + kk * 32 + hf * 8;
          vf[d4].q[0] = *(const v4u*)&Vs[vb];
          vf[d4].q[1] = *(const v4u*)&Vs[vb + 16];
        }
#pragma unroll
        for (int d4 = 0; d4 < 4; ++d4)
          o[dg * 4 + d4].v = __builtin_amdgcn_wmma_f32_16x16x32_bf16(
              false, pa.v, false, vf[d4].v, (short)0, o[dg * 4 + d4].v,
              false, false);
      }
    }
  }

  // normalize and store to (B,S,D) bf16
  const int b = bh >> 4, h = bh & 15;
#pragma unroll
  for (int r = 0; r < 8; ++r) lr[r] = 1.0f / lr[r];
#pragma unroll
  for (int d = 0; d < 8; ++d) {
    int hd = d * 16 + m;
#pragma unroll
    for (int r = 0; r < 8; ++r) {
      int s = qrow0 + hf * 8 + r;
      O[((size_t)b * S_ + s) * D_ + h * HD_ + hd] = f2bf(o[d].f[r] * lr[r]);
    }
  }
}

// ---------------------------------------------------------------- launch
extern "C" void kernel_launch(void* const* d_in, const int* in_sizes, int n_in,
                              void* d_out, int out_size, void* d_ws,
                              size_t ws_size, hipStream_t stream) {
  const float* q  = (const float*)d_in[0];
  const float* k  = (const float*)d_in[1];
  const float* v  = (const float*)d_in[2];
  // d_in[3] = attention_mask (causal, applied analytically)
  const float* wq = (const float*)d_in[4];
  const float* bq = (const float*)d_in[5];
  const float* wk = (const float*)d_in[6];
  const float* bk = (const float*)d_in[7];
  const float* wv = (const float*)d_in[8];
  const float* bv = (const float*)d_in[9];
  const float* wo = (const float*)d_in[10];
  const float* bo = (const float*)d_in[11];
  float* out = (float*)d_out;

  unsigned short* ws = (unsigned short*)d_ws;
  const size_t DDe = (size_t)D_ * D_;   // 4.19M elems
  const size_t NDe = (size_t)N_ * D_;   // 8.39M elems
  unsigned short* wo_b = ws;            // keep wo first (never overlapped)
  unsigned short* wq_b = wo_b + DDe;
  unsigned short* wk_b = wq_b + DDe;
  unsigned short* wv_b = wk_b + DDe;
  unsigned short* Qb   = wv_b + DDe;
  unsigned short* Kb   = Qb + NDe;
  unsigned short* Vtb  = Kb + NDe;
  unsigned short* AOb  = wq_b;          // attn out overlaps wq_b+wk_b (dead)

  // convert weights to bf16 once
  int n4 = (int)(DDe / 4);
  dim3 cg((n4 + 255) / 256);
  cvt_f32_bf16<<<cg, 256, 0, stream>>>(wo, wo_b, n4);
  cvt_f32_bf16<<<cg, 256, 0, stream>>>(wq, wq_b, n4);
  cvt_f32_bf16<<<cg, 256, 0, stream>>>(wk, wk_b, n4);
  cvt_f32_bf16<<<cg, 256, 0, stream>>>(wv, wv_b, n4);

  dim3 gg(N_ / 128, D_ / 128);  // 32 x 16 blocks
  gemm_wmma<0, false><<<gg, 256, 0, stream>>>(q, nullptr, wq_b, bq, Qb, nullptr);
  gemm_wmma<0, false><<<gg, 256, 0, stream>>>(k, nullptr, wk_b, bk, Kb, nullptr);
  gemm_wmma<1, false><<<gg, 256, 0, stream>>>(v, nullptr, wv_b, bv, Vtb, nullptr);

  flash_wmma<<<dim3(S_ / 128, B_ * H_), 256, 0, stream>>>(Qb, Kb, Vtb, AOb);

  gemm_wmma<2, true><<<gg, 256, 0, stream>>>(nullptr, AOb, wo_b, bo, nullptr, out);
}
